// ViT_82575041233468
// MI455X (gfx1250) — matrix operand order repaired and checked by reading, not compiled, not run on hardware
//
#include <hip/hip_runtime.h>
#include <hip/hip_bf16.h>

// ---------------------------------------------------------------------------
// ViT forward for MI455X (gfx1250, wave32, WMMA).
// All GEMMs run on v_wmma_f32_16x16x32_f16 with f32 accumulation.
// ---------------------------------------------------------------------------

typedef __attribute__((ext_vector_type(16))) _Float16 v16h;
typedef __attribute__((ext_vector_type(8)))  float    v8f;
typedef __attribute__((ext_vector_type(4)))  float    f4raw;   // trivially-copyable 16B chunk

#define BATCH    128
#define S_LEN    197
#define D_MODEL  256
#define NHEAD    8
#define DHEAD    32
#define NLAYER   4
#define NPATCH   196
#define IN_DIM   768
#define OUTC     1000
#define MROWS    (BATCH * S_LEN)        // 25216, multiple of 128
#define MPAT     (BATCH * NPATCH)       // 25088, multiple of 128
#define ST       224                    // padded seq length for P*V K-loop

// ---------------- WMMA fragment helpers (CDNA5 ISA 7.12.2 layouts) ----------

__device__ inline v8f zero8() {
    v8f z;
#pragma unroll
    for (int i = 0; i < 8; ++i) z[i] = 0.f;
    return z;
}

__device__ inline v16h zero16h() {
    v16h z;
#pragma unroll
    for (int i = 0; i < 16; ++i) z[i] = (_Float16)0;
    return z;
}

// A-matrix 16x32 f16: lane r=lane&15 holds row r; hi=lane>>4 selects K groups:
// halves j -> K = 16*hi + j (contiguous, matching B)
__device__ inline v16h load_a_frag(const _Float16* rowk0, int hi) {
    union { f4raw q[2]; v16h v; } u;
    u.q[0] = *reinterpret_cast<const f4raw*>(rowk0 + 16 * hi);
    u.q[1] = *reinterpret_cast<const f4raw*>(rowk0 + 16 * hi + 8);
    return u.v;
}

// B-matrix 32x16 f16: lane holds column n = lane&15; halves j -> K = 16*hi + j
// (16 contiguous K values of one weight row stored (N,K) row-major)
__device__ inline v16h load_b_frag(const _Float16* nrowk0, int hi) {
    union { f4raw q[2]; v16h v; } u;
    const f4raw* p = reinterpret_cast<const f4raw*>(nrowk0 + 16 * hi);
    u.q[0] = p[0];
    u.q[1] = p[1];
    return u.v;
}

__device__ inline v8f wmma_f16(v16h a, v16h b, v8f c) {
    return __builtin_amdgcn_wmma_f32_16x16x32_f16(false, a, false, b, (short)0, c,
                                                  false, false);
}

// ---------------- generic WMMA GEMM:  out[m,n] = sum_k A[m,k] * W[n,k] ------
// block = 128 threads = 4 waves; each wave computes a 32(M) x 64(N) tile:
// 8 WMMAs per 32-wide k-step, each B fragment reused for two A fragments.
// grid.x = M/128, grid.y = ceil(N/64).  Epilogue: +bias, optional exact GELU,
// optional residual add (f32), writes f32 and/or f16 outputs (ld == N).
// NGUARD=false removes all N bounds checks (N multiple of 64).

template <bool NGUARD, bool GELU>
__global__ __launch_bounds__(128)
void gemm_wmma_kernel(const _Float16* __restrict__ A, const _Float16* __restrict__ W,
                      const float* __restrict__ bias, float* __restrict__ outF32,
                      _Float16* __restrict__ outF16, const float* __restrict__ resid,
                      int M, int N, int K)
{
    const int lane  = threadIdx.x & 31;
    const int wave  = threadIdx.x >> 5;
    const int r     = lane & 15;
    const int hi    = lane >> 4;
    const int mTile = blockIdx.x * 8 + wave * 2;   // two 16-row tiles per wave
    const int nBlk  = blockIdx.y * 64;

    v8f acc[2][4];
#pragma unroll
    for (int mt = 0; mt < 2; ++mt)
#pragma unroll
        for (int nt = 0; nt < 4; ++nt) acc[mt][nt] = zero8();

    const _Float16* arow0 = A + ((size_t)(mTile * 16 + r)) * K;
    const _Float16* arow1 = arow0 + (size_t)16 * K;

    for (int k0 = 0; k0 < K; k0 += 32) {
        __builtin_prefetch(arow0 + k0 + 128, 0, 1);      // global_prefetch_b8
        __builtin_prefetch(arow1 + k0 + 128, 0, 1);
        v16h a0 = load_a_frag(arow0 + k0, hi);
        v16h a1 = load_a_frag(arow1 + k0, hi);
#pragma unroll
        for (int nt = 0; nt < 4; ++nt) {
            int n = nBlk + nt * 16 + r;
            v16h b;
            if (NGUARD)
                b = (n < N) ? load_b_frag(W + (size_t)n * K + k0, hi) : zero16h();
            else
                b = load_b_frag(W + (size_t)n * K + k0, hi);
            acc[0][nt] = wmma_f16(a0, b, acc[0][nt]);
            acc[1][nt] = wmma_f16(a1, b, acc[1][nt]);
        }
    }

#pragma unroll
    for (int mt = 0; mt < 2; ++mt)
#pragma unroll
        for (int nt = 0; nt < 4; ++nt) {
            int col = nBlk + nt * 16 + r;
            if (NGUARD && col >= N) continue;
            float bv = bias ? bias[col] : 0.f;
#pragma unroll
            for (int i = 0; i < 8; ++i) {
                int row  = (mTile + mt) * 16 + i + 8 * hi;
                float v  = acc[mt][nt][i] + bv;
                if (GELU) v = 0.5f * v * (1.f + erff(v * 0.70710678118654752f));
                size_t idx = (size_t)row * N + col;
                if (resid)  v += resid[idx];
                if (outF32) outF32[idx] = v;
                if (outF16) outF16[idx] = (_Float16)v;
            }
        }
}

// ---------------- per-head QKV projection: K = DH = 32 -> single WMMA -------
// grid = (M/64, 2, H); out layout [M, H, 32] f16.

__global__ __launch_bounds__(128)
void qkv_proj_kernel(const _Float16* __restrict__ X16, const _Float16* __restrict__ Wh,
                     const float* __restrict__ bh, _Float16* __restrict__ out, int M)
{
    const int lane  = threadIdx.x & 31;
    const int wave  = threadIdx.x >> 5;
    const int r     = lane & 15;
    const int hi    = lane >> 4;
    const int mTile = blockIdx.x * 4 + wave;
    const int nt    = blockIdx.y;     // 0/1 -> output halves
    const int h     = blockIdx.z;
    const int m     = mTile * 16 + r;
    const int n     = nt * 16 + r;

    v16h a = load_a_frag(X16 + (size_t)m * D_MODEL + h * DHEAD, hi);
    v16h b = load_b_frag(Wh + ((size_t)h * DHEAD + n) * DHEAD, hi);
    v8f  c = wmma_f16(a, b, zero8());

    float bv = bh[h * DHEAD + n];
#pragma unroll
    for (int i = 0; i < 8; ++i) {
        int row = mTile * 16 + i + 8 * hi;
        out[((size_t)row * NHEAD + h) * DHEAD + n] = (_Float16)(c[i] + bv);
    }
}

// ---------------- fused attention: one workgroup per (b,h) ------------------
// LDS: K,V tiles (224x32 f16) + per-wave probability tile (16x224 f16) = 56KB.

__global__ __launch_bounds__(128)
void attention_kernel(const _Float16* __restrict__ Qb, const _Float16* __restrict__ Kb,
                      const _Float16* __restrict__ Vb, _Float16* __restrict__ Ob)
{
    __shared__ __align__(16) _Float16 Ksh[ST * DHEAD];
    __shared__ __align__(16) _Float16 Vsh[ST * DHEAD];
    __shared__ __align__(16) _Float16 Psh[4][16 * ST];

    const int bh  = blockIdx.x;
    const int b   = bh / NHEAD;
    const int h   = bh % NHEAD;
    const int tid = threadIdx.x;

    for (int idx = tid; idx < ST * DHEAD; idx += 128) {
        int t = idx / DHEAD, e = idx % DHEAD;
        _Float16 kv = (_Float16)0, vv = (_Float16)0;
        if (t < S_LEN) {
            size_t g = (((size_t)b * S_LEN + t) * NHEAD + h) * DHEAD + e;
            kv = Kb[g];
            vv = Vb[g];
        }
        Ksh[idx] = kv;
        Vsh[idx] = vv;
    }
    __syncthreads();

    const int lane = tid & 31, wave = tid >> 5;
    const int r = lane & 15, hi = lane >> 4;
    const float scale = 0.17677669529663687f;   // 1/sqrt(32)

    for (int qt = wave; qt < 13; qt += 4) {     // 13 q-row tiles of 16 cover S=197
        int s  = qt * 16 + r;
        int sA = (s < S_LEN) ? s : (S_LEN - 1);
        v16h qa = load_a_frag(Qb + (((size_t)b * S_LEN + sA) * NHEAD + h) * DHEAD, hi);

        // scores = Q(16x32) x K^T(32x208), one WMMA per 16-col tile
        v8f sc[13];
#pragma unroll
        for (int tt = 0; tt < 13; ++tt) {
            v16h kb = load_b_frag(Ksh + (tt * 16 + r) * DHEAD, hi);
            sc[tt] = wmma_f16(qa, kb, zero8());
        }

        // row-wise softmax: C-layout row = i + 8*hi, col = tt*16 + (lane&15)
#pragma unroll
        for (int i = 0; i < 8; ++i) {
            float mx = -1e30f;
#pragma unroll
            for (int tt = 0; tt < 13; ++tt) {
                int col = tt * 16 + r;
                float v = (col < S_LEN) ? sc[tt][i] * scale : -1e30f;
                sc[tt][i] = v;
                mx = fmaxf(mx, v);
            }
            for (int off = 1; off < 16; off <<= 1) mx = fmaxf(mx, __shfl_xor(mx, off, 32));
            float sum = 0.f;
#pragma unroll
            for (int tt = 0; tt < 13; ++tt) {
                int col = tt * 16 + r;
                float p = (col < S_LEN) ? __expf(sc[tt][i] - mx) : 0.f;
                sc[tt][i] = p;
                sum += p;
            }
            for (int off = 1; off < 16; off <<= 1) sum += __shfl_xor(sum, off, 32);
            float inv = 1.f / sum;
#pragma unroll
            for (int tt = 0; tt < 13; ++tt) sc[tt][i] *= inv;
        }

        // restage probabilities (A-layout needs f16, different striping than C)
        _Float16* P = &Psh[wave][0];
#pragma unroll
        for (int i = 0; i < 8; ++i) {
            int row = i + 8 * hi;
#pragma unroll
            for (int tt = 0; tt < 13; ++tt)
                P[row * ST + tt * 16 + r] = (_Float16)sc[tt][i];
            P[row * ST + 208 + r] = (_Float16)0;   // zero pad cols 208..223
        }
        asm volatile("s_wait_dscnt 0" ::: "memory");   // intra-wave LDS RAW
        __builtin_amdgcn_wave_barrier();

        // O = P(16x224) x V(224x32)
        v8f oa[2];
        oa[0] = zero8();
        oa[1] = zero8();
        for (int k0 = 0; k0 < ST; k0 += 32) {
            v16h pa = load_a_frag(P + r * ST + k0, hi);
#pragma unroll
            for (int nt = 0; nt < 2; ++nt) {
                v16h vb;
#pragma unroll
                for (int j = 0; j < 16; ++j)
                    vb[j] = Vsh[(k0 + 16 * hi + j) * DHEAD + nt * 16 + r];
                oa[nt] = wmma_f16(pa, vb, oa[nt]);
            }
        }
#pragma unroll
        for (int nt = 0; nt < 2; ++nt)
#pragma unroll
            for (int i = 0; i < 8; ++i) {
                int row = qt * 16 + i + 8 * hi;
                if (row < S_LEN)
                    Ob[(((size_t)b * S_LEN + row) * NHEAD + h) * DHEAD + nt * 16 + r] =
                        (_Float16)oa[nt][i];
            }
    }
}

// ---------------- elementwise / reduction kernels ---------------------------

__global__ void im2col_kernel(const float* __restrict__ img, _Float16* __restrict__ Pm)
{
    size_t idx = (size_t)blockIdx.x * 256 + threadIdx.x;
    const size_t total = (size_t)MPAT * IN_DIM;
    if (idx >= total) return;
    int i  = (int)(idx % IN_DIM);
    size_t bp = idx / IN_DIM;
    int p  = (int)(bp % NPATCH);
    int b  = (int)(bp / NPATCH);
    int c  = i / 256, rem = i % 256;
    int dy = rem / 16, dx = rem % 16;
    int py = p / 14,  px = p % 14;
    int yy = py * 16 + dy, xx = px * 16 + dx;
    Pm[idx] = (_Float16)img[(((size_t)b * 3 + c) * 224 + yy) * 224 + xx];
}

__global__ void cast_f32_f16_kernel(const float* __restrict__ src,
                                    _Float16* __restrict__ dst, int n)
{
    int idx = blockIdx.x * 256 + threadIdx.x;
    if (idx < n) dst[idx] = (_Float16)src[idx];
}

__global__ void assemble_kernel(const float* __restrict__ Tok,
                                const float* __restrict__ cls, float* __restrict__ X)
{
    size_t idx = (size_t)blockIdx.x * 256 + threadIdx.x;   // B*S*D total
    int d = (int)(idx % D_MODEL);
    size_t bs = idx / D_MODEL;
    int s = (int)(bs % S_LEN);
    int b = (int)(bs / S_LEN);
    float v = (s == 0) ? cls[d]
                       : Tok[((size_t)b * NPATCH + (s - 1)) * D_MODEL + d];
    float expo  = (float)(2 * (d / 2)) / (float)D_MODEL;
    float angle = (float)s / __powf(10000.f, expo);
    v += (d & 1) ? __cosf(angle) : __sinf(angle);
    X[idx] = v;
}

__global__ __launch_bounds__(256)
void layernorm_kernel(const float* __restrict__ X, const float* __restrict__ g,
                      const float* __restrict__ bta, _Float16* __restrict__ out)
{
    __shared__ float red[256];
    const int row = blockIdx.x, tid = threadIdx.x;
    float x = X[(size_t)row * D_MODEL + tid];
    red[tid] = x;
    __syncthreads();
    for (int s2 = 128; s2 > 0; s2 >>= 1) {
        if (tid < s2) red[tid] += red[tid + s2];
        __syncthreads();
    }
    float mu = red[0] / 256.f;
    __syncthreads();
    float d = x - mu;
    red[tid] = d * d;
    __syncthreads();
    for (int s2 = 128; s2 > 0; s2 >>= 1) {
        if (tid < s2) red[tid] += red[tid + s2];
        __syncthreads();
    }
    float var = red[0] / 256.f;
    float y = d * rsqrtf(var + 1e-5f) * g[tid] + bta[tid];
    out[(size_t)row * D_MODEL + tid] = (_Float16)y;
}

__global__ void gather_cls_kernel(const float* __restrict__ X, _Float16* __restrict__ out)
{
    int idx = blockIdx.x * 256 + threadIdx.x;   // 128*256
    int b = idx / D_MODEL, d = idx % D_MODEL;
    out[idx] = (_Float16)X[((size_t)b * S_LEN) * D_MODEL + d];
}

__global__ __launch_bounds__(256)
void softmax_kernel(const float* __restrict__ logits, float* __restrict__ out)
{
    __shared__ float red[256];
    const int row = blockIdx.x, tid = threadIdx.x;
    float mx = -1e30f;
    for (int c = tid; c < OUTC; c += 256)
        mx = fmaxf(mx, logits[(size_t)row * OUTC + c]);
    red[tid] = mx;
    __syncthreads();
    for (int s2 = 128; s2 > 0; s2 >>= 1) {
        if (tid < s2) red[tid] = fmaxf(red[tid], red[tid + s2]);
        __syncthreads();
    }
    mx = red[0];
    __syncthreads();
    float sum = 0.f;
    for (int c = tid; c < OUTC; c += 256)
        sum += __expf(logits[(size_t)row * OUTC + c] - mx);
    red[tid] = sum;
    __syncthreads();
    for (int s2 = 128; s2 > 0; s2 >>= 1) {
        if (tid < s2) red[tid] += red[tid + s2];
        __syncthreads();
    }
    float inv = 1.f / red[0];
    for (int c = tid; c < OUTC; c += 256)
        out[(size_t)row * OUTC + c] = __expf(logits[(size_t)row * OUTC + c] - mx) * inv;
}

// ---------------- host orchestration ---------------------------------------

extern "C" void kernel_launch(void* const* d_in, const int* in_sizes, int n_in,
                              void* d_out, int out_size, void* d_ws, size_t ws_size,
                              hipStream_t stream)
{
    (void)in_sizes; (void)n_in; (void)out_size; (void)ws_size;

    const float* images  = (const float*)d_in[0];
    const float* w_map   = (const float*)d_in[1];
    const float* b_map   = (const float*)d_in[2];
    const float* cls_tok = (const float*)d_in[3];
    const float* norm1_g = (const float*)d_in[4];
    const float* norm1_b = (const float*)d_in[5];
    const float* wq      = (const float*)d_in[6];
    const float* bq      = (const float*)d_in[7];
    const float* wk      = (const float*)d_in[8];
    const float* bk      = (const float*)d_in[9];
    const float* wv      = (const float*)d_in[10];
    const float* bv      = (const float*)d_in[11];
    const float* w_last  = (const float*)d_in[12];
    const float* b_last  = (const float*)d_in[13];
    const float* norm2_g = (const float*)d_in[14];
    const float* norm2_b = (const float*)d_in[15];
    const float* w_mlp1  = (const float*)d_in[16];
    const float* b_mlp1  = (const float*)d_in[17];
    const float* w_mlp2  = (const float*)d_in[18];
    const float* b_mlp2  = (const float*)d_in[19];
    const float* w_out   = (const float*)d_in[20];
    const float* b_out   = (const float*)d_in[21];

    char* ws = (char*)d_ws;
    size_t off = 0;
    auto alloc = [&](size_t bytes) -> size_t {
        off = (off + 255) & ~(size_t)255;
        size_t r = off;
        off += bytes;
        return r;
    };

    const size_t pmat_o   = alloc((size_t)MPAT * IN_DIM * 2);    // dead after embed
    const size_t tok_o    = alloc((size_t)MPAT * D_MODEL * 4);   // dead after embed
    const size_t X_o      = alloc((size_t)MROWS * D_MODEL * 4);
    const size_t X16_o    = alloc((size_t)MROWS * D_MODEL * 2);
    const size_t Q_o      = alloc((size_t)MROWS * D_MODEL * 2);
    const size_t K_o      = alloc((size_t)MROWS * D_MODEL * 2);
    const size_t V_o      = alloc((size_t)MROWS * D_MODEL * 2);
    const size_t O_o      = alloc((size_t)MROWS * D_MODEL * 2);
    const size_t logit_o  = alloc((size_t)BATCH * OUTC * 4);
    const size_t cls16_o  = alloc((size_t)BATCH * D_MODEL * 2);
    const size_t wmap16_o = alloc((size_t)D_MODEL * IN_DIM * 2);
    const size_t wq16_o   = alloc((size_t)NLAYER * NHEAD * DHEAD * DHEAD * 2);
    const size_t wk16_o   = alloc((size_t)NLAYER * NHEAD * DHEAD * DHEAD * 2);
    const size_t wv16_o   = alloc((size_t)NLAYER * NHEAD * DHEAD * DHEAD * 2);
    const size_t wlast16_o= alloc((size_t)NLAYER * D_MODEL * D_MODEL * 2);
    const size_t wmlp1_o  = alloc((size_t)NLAYER * 4 * D_MODEL * D_MODEL * 2);
    const size_t wmlp2_o  = alloc((size_t)NLAYER * 4 * D_MODEL * D_MODEL * 2);
    const size_t wout16_o = alloc((size_t)OUTC * D_MODEL * 2);

    _Float16* Pmat   = (_Float16*)(ws + pmat_o);
    float*    Tok    = (float*)(ws + tok_o);
    float*    X      = (float*)(ws + X_o);
    _Float16* X16    = (_Float16*)(ws + X16_o);
    _Float16* Qb     = (_Float16*)(ws + Q_o);
    _Float16* Kb     = (_Float16*)(ws + K_o);
    _Float16* Vb     = (_Float16*)(ws + V_o);
    _Float16* Ob     = (_Float16*)(ws + O_o);
    float*    Logits = (float*)(ws + logit_o);
    _Float16* Cls16  = (_Float16*)(ws + cls16_o);
    _Float16* Wmap16 = (_Float16*)(ws + wmap16_o);
    _Float16* Wq16   = (_Float16*)(ws + wq16_o);
    _Float16* Wk16   = (_Float16*)(ws + wk16_o);
    _Float16* Wv16   = (_Float16*)(ws + wv16_o);
    _Float16* Wlast16= (_Float16*)(ws + wlast16_o);
    _Float16* Wmlp1  = (_Float16*)(ws + wmlp1_o);
    _Float16* Wmlp2  = (_Float16*)(ws + wmlp2_o);
    _Float16* Wout16 = (_Float16*)(ws + wout16_o);
    // H1 (25216x1024 f16, 51.6MB) aliases Pmat+Tok (64.2MB), both dead by then.
    _Float16* H1     = (_Float16*)(ws + pmat_o);

    auto cast = [&](const float* s, _Float16* dst, int n) {
        cast_f32_f16_kernel<<<(n + 255) / 256, 256, 0, stream>>>(s, dst, n);
    };
    cast(w_map,  Wmap16, D_MODEL * IN_DIM);
    cast(wq,     Wq16,   NLAYER * NHEAD * DHEAD * DHEAD);
    cast(wk,     Wk16,   NLAYER * NHEAD * DHEAD * DHEAD);
    cast(wv,     Wv16,   NLAYER * NHEAD * DHEAD * DHEAD);
    cast(w_last, Wlast16,NLAYER * D_MODEL * D_MODEL);
    cast(w_mlp1, Wmlp1,  NLAYER * 4 * D_MODEL * D_MODEL);
    cast(w_mlp2, Wmlp2,  NLAYER * 4 * D_MODEL * D_MODEL);
    cast(w_out,  Wout16, OUTC * D_MODEL);

    // patch embed
    {
        const size_t total = (size_t)MPAT * IN_DIM;
        im2col_kernel<<<(unsigned)((total + 255) / 256), 256, 0, stream>>>(images, Pmat);
        gemm_wmma_kernel<false, false><<<dim3(MPAT / 128, D_MODEL / 64), 128, 0, stream>>>(
            Pmat, Wmap16, b_map, Tok, nullptr, nullptr, MPAT, D_MODEL, IN_DIM);
        assemble_kernel<<<MROWS, 256, 0, stream>>>(Tok, cls_tok, X);
    }

    for (int l = 0; l < NLAYER; ++l) {
        layernorm_kernel<<<MROWS, 256, 0, stream>>>(X, norm1_g + l * D_MODEL,
                                                    norm1_b + l * D_MODEL, X16);
        const size_t wsz = (size_t)NHEAD * DHEAD * DHEAD;   // per-layer head weights
        qkv_proj_kernel<<<dim3(MROWS / 64, 2, NHEAD), 128, 0, stream>>>(
            X16, Wq16 + l * wsz, bq + l * NHEAD * DHEAD, Qb, MROWS);
        qkv_proj_kernel<<<dim3(MROWS / 64, 2, NHEAD), 128, 0, stream>>>(
            X16, Wk16 + l * wsz, bk + l * NHEAD * DHEAD, Kb, MROWS);
        qkv_proj_kernel<<<dim3(MROWS / 64, 2, NHEAD), 128, 0, stream>>>(
            X16, Wv16 + l * wsz, bv + l * NHEAD * DHEAD, Vb, MROWS);

        attention_kernel<<<BATCH * NHEAD, 128, 0, stream>>>(Qb, Kb, Vb, Ob);

        gemm_wmma_kernel<false, false><<<dim3(MROWS / 128, D_MODEL / 64), 128, 0, stream>>>(
            Ob, Wlast16 + (size_t)l * D_MODEL * D_MODEL, b_last + l * D_MODEL,
            X, nullptr, X, MROWS, D_MODEL, D_MODEL);

        layernorm_kernel<<<MROWS, 256, 0, stream>>>(X, norm2_g + l * D_MODEL,
                                                    norm2_b + l * D_MODEL, X16);
        gemm_wmma_kernel<false, true><<<dim3(MROWS / 128, (4 * D_MODEL) / 64), 128, 0, stream>>>(
            X16, Wmlp1 + (size_t)l * 4 * D_MODEL * D_MODEL, b_mlp1 + l * 4 * D_MODEL,
            nullptr, H1, nullptr, MROWS, 4 * D_MODEL, D_MODEL);
        gemm_wmma_kernel<false, false><<<dim3(MROWS / 128, D_MODEL / 64), 128, 0, stream>>>(
            H1, Wmlp2 + (size_t)l * 4 * D_MODEL * D_MODEL, b_mlp2 + l * D_MODEL,
            X, nullptr, X, MROWS, D_MODEL, 4 * D_MODEL);
    }

    gather_cls_kernel<<<BATCH, 256, 0, stream>>>(X, Cls16);
    gemm_wmma_kernel<true, false><<<dim3(BATCH / 128, (OUTC + 63) / 64), 128, 0, stream>>>(
        Cls16, Wout16, b_out, Logits, nullptr, nullptr, BATCH, OUTC, D_MODEL);
    softmax_kernel<<<BATCH, 256, 0, stream>>>(Logits, (float*)d_out);
}
